// TeacherDecoder_87625922773505
// MI455X (gfx1250) — compile-verified, matrix operand order checked by reading, not run on hardware
//
#include <hip/hip_runtime.h>
#include <hip/hip_bf16.h>
#include <cstdint>
#include <cstddef>

// Problem constants (reference: H=2048, V=50257, output_length=256).
#define H   2048
#define H4  8192          // 4*H gate rows
#define V   50257
#define T   256
#define REC_BLOCKS 32     // 32 blocks * 256 threads = 8192 threads = one gate row each
#define NT  32            // vocab columns per GEMM block
#define NKC (H / 32)      // number of K chunks

typedef __attribute__((ext_vector_type(16))) __bf16 v16bf;
typedef __attribute__((ext_vector_type(8)))  __bf16 v8bf;
typedef __attribute__((ext_vector_type(8)))  float  v8f;
typedef __attribute__((ext_vector_type(4)))  int    v4i;

typedef __attribute__((address_space(1))) v4i* g_v4i_ptr;   // global <4 x i32>*
typedef __attribute__((address_space(3))) v4i* l_v4i_ptr;   // LDS    <4 x i32>*

#if defined(__has_builtin)
#if __has_builtin(__builtin_amdgcn_global_load_async_to_lds_b128)
#define HAS_ASYNC_LDS 1
#endif
#endif
#ifndef HAS_ASYNC_LDS
#define HAS_ASYNC_LDS 0
#endif

__device__ __forceinline__ float sigmoid_f(float x) {
    return 1.0f / (1.0f + __expf(-x));
}

__device__ __forceinline__ v16bf cat8(v8bf lo, v8bf hi) {
    return __builtin_shufflevector(lo, hi, 0, 1, 2, 3, 4, 5, 6, 7,
                                           8, 9, 10, 11, 12, 13, 14, 15);
}

// Phase-counting grid barrier; counter zeroed by hipMemsetAsync pre-launch.
__device__ __forceinline__ void grid_sync(int* counter, int target) {
    __syncthreads();
    if (threadIdx.x == 0) {
        __threadfence();
        __hip_atomic_fetch_add(counter, 1, __ATOMIC_RELEASE, __HIP_MEMORY_SCOPE_AGENT);
        while (__hip_atomic_load(counter, __ATOMIC_ACQUIRE, __HIP_MEMORY_SCOPE_AGENT) < target) {
            __builtin_amdgcn_s_sleep(1);
        }
    }
    __syncthreads();
    __threadfence();
}

// ---------------------------------------------------------------------------
// Phase 1: persistent LSTM recurrence. w_hh (67 MB) is L2-resident (192 MB L2),
// so the 256 re-reads are L2 traffic. One gate row per thread; h staged in LDS.
// Emits h_t fp32 state and bf16 h_all laid out [T][H].
// ---------------------------------------------------------------------------
__global__ void __launch_bounds__(256)
lstm_recurrence_kernel(const float* __restrict__ h0, const float* __restrict__ c0,
                       const float* __restrict__ w_hh,
                       const float* __restrict__ b_ih, const float* __restrict__ b_hh,
                       float* hA, float* cA, float* gates,
                       __bf16* hT, int* barrier)
{
    __shared__ float hs[H];                       // 8 KB staging of current h
    const int tid  = threadIdx.x;
    const int gtid = blockIdx.x * 256 + tid;      // 0..8191 -> gate row
    int phase = 0;

    if (gtid < H) { hA[gtid] = h0[gtid]; cA[gtid] = c0[gtid]; }
    __threadfence();
    grid_sync(barrier, REC_BLOCKS * ++phase);

    const float gate_bias = b_ih[gtid] + b_hh[gtid];    // w_ih @ 0 == 0
    const float4* wrow = (const float4*)(w_hh + (size_t)gtid * H);

    for (int t = 0; t < T; ++t) {
        {
            float4*       dst = (float4*)hs;
            const float4* src = (const float4*)hA;
            for (int i = tid; i < H / 4; i += 256) dst[i] = src[i];
        }
        __syncthreads();

        float acc = 0.0f;
        const float4* hv4 = (const float4*)hs;
        #pragma unroll 4
        for (int k = 0; k < H / 4; ++k) {
            float4 w = wrow[k];
            float4 h = hv4[k];
            acc = fmaf(w.x, h.x, acc);
            acc = fmaf(w.y, h.y, acc);
            acc = fmaf(w.z, h.z, acc);
            acc = fmaf(w.w, h.w, acc);
        }
        gates[gtid] = acc + gate_bias;
        grid_sync(barrier, REC_BLOCKS * ++phase);

        if (gtid < H) {
            const int j = gtid;
            float gi = sigmoid_f(gates[j]);
            float gf = sigmoid_f(gates[H     + j]);
            float gg = tanhf   (gates[2 * H + j]);
            float go = sigmoid_f(gates[3 * H + j]);
            float c  = gf * cA[j] + gi * gg;
            float h  = go * tanhf(c);
            cA[j] = c;
            hA[j] = h;
            hT[(size_t)t * H + j] = (__bf16)h;    // [T][H], coalesced
        }
        __threadfence();
        grid_sync(barrier, REC_BLOCKS * ++phase);
    }
}

// ---------------------------------------------------------------------------
// Phase 2: logits[t][v] = sum_k h_all[t][k] * W_out[v][k] + b_out[v].
// M = T = 256 (whole time axis), N = NT = 32 vocab cols, K chunk = 32.
// Double-buffered LDS, one barrier per K chunk. The A tile (already bf16 in
// workspace) is staged with GLOBAL_LOAD_ASYNC_TO_LDS_B128 (ASYNCcnt), ordered
// with s_wait_asynccnt; the B tile (needs fp32->bf16) is loaded into registers
// one chunk ahead so its loadcnt wait hides behind a chunk of WMMAs.
// ---------------------------------------------------------------------------
__global__ void __launch_bounds__(256)
logits_gemm_kernel(const float* __restrict__ W_out, const float* __restrict__ b_out,
                   const __bf16* __restrict__ hT, float* __restrict__ out /* [T][V] */)
{
    __shared__ __align__(16) __bf16 As[2][T][40];   // 2 x (256 x 32 + pad), 40 KB
    __shared__ __align__(16) __bf16 Bs[2][NT][40];  // 2 x (32 x 32 + pad), 5 KB

    const int tid  = threadIdx.x;
    const int lane = tid & 31;
    const int wave = tid >> 5;              // 0..7
    const int v0   = blockIdx.x * NT;

    const int kb   = (lane >> 4) * 8;       // K base: 0 (lanes 0-15) / 8 (16-31)
    const int n    = lane & 15;             // N within subtile
    const int row0 = wave * 32 + (lane & 15);   // A row, M-subtile 0
    const int row1 = row0 + 16;                 // A row, M-subtile 1

    // B staging coordinates: 32 rows x 32 K, 4 floats per thread.
    const int bn = tid >> 3;                // 0..31
    const int bk = (tid & 7) * 4;           // 0,4,...,28
    const int bv = v0 + bn;
    const float* bsrc = (bv < V) ? (W_out + (size_t)bv * H + bk) : nullptr;

    // A staging: one 64B row slice per thread.
    const __bf16* asrc = hT + (size_t)tid * H;

    auto stage_a = [&](int buf, int kk) {
#if HAS_ASYNC_LDS
        const __bf16* g = asrc + kk;
        #pragma unroll
        for (int q = 0; q < 4; ++q) {
            __builtin_amdgcn_global_load_async_to_lds_b128(
                (g_v4i_ptr)(g + q * 8),
                (l_v4i_ptr)&As[buf][tid][q * 8],
                0, 0);
        }
#else
        const uint4* src = (const uint4*)(asrc + kk);
        uint4* dst = (uint4*)&As[buf][tid][0];
        dst[0] = src[0]; dst[1] = src[1]; dst[2] = src[2]; dst[3] = src[3];
#endif
    };

    v8f acc00 = {}, acc01 = {}, acc10 = {}, acc11 = {};

    // ---- prologue: stage K-chunk 0 ----
    float4 breg = make_float4(0.f, 0.f, 0.f, 0.f);
    if (bsrc) breg = *(const float4*)bsrc;
    stage_a(0, 0);

    for (int i = 0; i < NKC; ++i) {
        const int cur = i & 1;
        const int nxt = cur ^ 1;

        // Commit the prefetched B chunk (cvt fp32->bf16) into Bs[cur].
        Bs[cur][bn][bk]     = (__bf16)breg.x;
        Bs[cur][bn][bk + 1] = (__bf16)breg.y;
        Bs[cur][bn][bk + 2] = (__bf16)breg.z;
        Bs[cur][bn][bk + 3] = (__bf16)breg.w;

#if HAS_ASYNC_LDS
        asm volatile("s_wait_asynccnt 0x0" ::: "memory");  // A[cur] landed in LDS
#endif
        __syncthreads();    // buf[cur] fully staged; all waves past buf[nxt] reads

        // Kick off next chunk's staging; it overlaps the WMMAs below.
        if (i + 1 < NKC) {
            const int kk2 = (i + 1) * 32;
            stage_a(nxt, kk2);
            if (bsrc) breg = *(const float4*)(bsrc + kk2);
        }

        // Fragments: two contiguous 16B runs per lane -> ds_load_b128 pairs.
        v16bf a0 = cat8(*(const v8bf*)&As[cur][row0][kb], *(const v8bf*)&As[cur][row0][kb + 16]);
        v16bf a1 = cat8(*(const v8bf*)&As[cur][row1][kb], *(const v8bf*)&As[cur][row1][kb + 16]);
        v16bf b0 = cat8(*(const v8bf*)&Bs[cur][n][kb],      *(const v8bf*)&Bs[cur][n][kb + 16]);
        v16bf b1 = cat8(*(const v8bf*)&Bs[cur][n + 16][kb], *(const v8bf*)&Bs[cur][n + 16][kb + 16]);

        acc00 = __builtin_amdgcn_wmma_f32_16x16x32_bf16(false, a0, false, b0, (short)0, acc00, false, false);
        acc01 = __builtin_amdgcn_wmma_f32_16x16x32_bf16(false, a0, false, b1, (short)0, acc01, false, false);
        acc10 = __builtin_amdgcn_wmma_f32_16x16x32_bf16(false, a1, false, b0, (short)0, acc10, false, false);
        acc11 = __builtin_amdgcn_wmma_f32_16x16x32_bf16(false, a1, false, b1, (short)0, acc11, false, false);
    }

    // Epilogue. C/D layout: VGPR r -> M = r + 8*(lane>>4); N = lane&15.
    // t rows are always in range; only the vocab column needs a per-lane
    // guard, so stores lower to coalesced b32 clauses.
    const int va  = v0 + n;
    const int vb  = va + 16;
    const int thi = 8 * (lane >> 4);
    const float ba = (va < V) ? b_out[va] : 0.0f;
    const float bb = (vb < V) ? b_out[vb] : 0.0f;
    #pragma unroll
    for (int r = 0; r < 8; ++r) {
        const int t0 = wave * 32 + thi + r;     // M-subtile 0
        const int t1 = t0 + 16;                 // M-subtile 1
        if (va < V) {
            out[(size_t)t0 * V + va] = acc00[r] + ba;
            out[(size_t)t1 * V + va] = acc10[r] + ba;
        }
        if (vb < V) {
            out[(size_t)t0 * V + vb] = acc01[r] + bb;
            out[(size_t)t1 * V + vb] = acc11[r] + bb;
        }
    }
}

// ---------------------------------------------------------------------------
// Phase 3: row-wise softmax over V, one block per timestep.
// ---------------------------------------------------------------------------
__global__ void __launch_bounds__(256)
softmax_rows_kernel(float* __restrict__ out)
{
    __shared__ float red[256];
    float* row = out + (size_t)blockIdx.x * V;
    const int tid = threadIdx.x;

    float m = -3.402823466e38f;
    for (int i = tid; i < V; i += 256) m = fmaxf(m, row[i]);
    red[tid] = m; __syncthreads();
    for (int s = 128; s > 0; s >>= 1) {
        if (tid < s) red[tid] = fmaxf(red[tid], red[tid + s]);
        __syncthreads();
    }
    m = red[0]; __syncthreads();

    float sum = 0.0f;
    for (int i = tid; i < V; i += 256) sum += __expf(row[i] - m);
    red[tid] = sum; __syncthreads();
    for (int s = 128; s > 0; s >>= 1) {
        if (tid < s) red[tid] += red[tid + s];
        __syncthreads();
    }
    const float inv = 1.0f / red[0];
    __syncthreads();

    for (int i = tid; i < V; i += 256) row[i] = __expf(row[i] - m) * inv;
}

extern "C" void kernel_launch(void* const* d_in, const int* in_sizes, int n_in,
                              void* d_out, int out_size, void* d_ws, size_t ws_size,
                              hipStream_t stream) {
    (void)in_sizes; (void)n_in; (void)out_size; (void)ws_size;

    const float* h0    = (const float*)d_in[0];
    const float* c0    = (const float*)d_in[1];
    /* d_in[2] = w_ih, unused: the fed embedding is all-zero, so w_ih@x == 0 */
    const float* w_hh  = (const float*)d_in[3];
    const float* b_ih  = (const float*)d_in[4];
    const float* b_hh  = (const float*)d_in[5];
    const float* W_out = (const float*)d_in[6];
    const float* b_out = (const float*)d_in[7];
    /* d_in[8] = output_length (== 256, baked in as T) */

    // Workspace layout (~1.1 MB): barrier | hA | cA | gates | hT(bf16 [T][H])
    char*   ws      = (char*)d_ws;
    int*    barrier = (int*)ws;                          // 256 B
    float*  hA      = (float*)(ws + 256);                // H fp32
    float*  cA      = hA + H;                            // H fp32
    float*  gates   = cA + H;                            // 4H fp32
    __bf16* hT      = (__bf16*)(gates + H4);             // T*H bf16 (1 MB)

    (void)hipMemsetAsync(barrier, 0, 256, stream);       // graph-capture safe

    lstm_recurrence_kernel<<<REC_BLOCKS, 256, 0, stream>>>(
        h0, c0, w_hh, b_ih, b_hh, hA, cA, gates, hT, barrier);

    logits_gemm_kernel<<<(V + NT - 1) / NT, 256, 0, stream>>>(
        W_out, b_out, hT, (float*)d_out);

    softmax_rows_kernel<<<T, 256, 0, stream>>>((float*)d_out);
}
